// fusion_36455682409119
// MI455X (gfx1250) — matrix operand order repaired and checked by reading, not compiled, not run on hardware
//
#include <hip/hip_runtime.h>

// ---------------------------------------------------------------------------
// EDVR PCD-alignment block on gfx1250 (MI455X).
// All conv / deformable-conv GEMMs run on v_wmma_f32_16x16x32_bf16 (wave32),
// f32 accumulate. im2col tiles staged in LDS as 96-wide K panels so each
// barrier pair covers 3 WMMA k-steps (6 WMMAs/wave) -> fewer barriers, more
// back-to-back XDL issue. bf16 weights pre-converted once into workspace.
// ---------------------------------------------------------------------------

#define B_   4
#define C_   64
#define H_   180
#define W_   320
#define HW_  (H_ * W_)
#define DG_  8
#define OM_  (3 * DG_ * 9)   // 216

#define TILE_N    64
#define KSTEP     32
#define KPANEL    96                       // 3 WMMA k-steps per LDS panel
#define BK_STRIDE 98                       // 196 B row stride = 49 dwords (odd -> conflict-free)
#define GRID_X  ((W_ / TILE_N) * H_ * B_)  // 5 * 180 * 4 = 3600

typedef __attribute__((ext_vector_type(16))) __bf16 v16bf;
typedef __attribute__((ext_vector_type(8)))  float  v8f;

union Frag16 {
    v16bf        v;
    unsigned int u[8];
};

__device__ __forceinline__ unsigned short f2bf(float f) {
    // fp32 -> bf16 round-to-nearest-even (bit manipulation; no __bf16 scalar math)
    unsigned int u = __float_as_uint(f);
    unsigned int r = u + 0x7FFFu + ((u >> 16) & 1u);
    return (unsigned short)(r >> 16);
}

__device__ __forceinline__ float ld_guard(const float* __restrict__ ch, int iy, int ix) {
    return (iy >= 0 && iy < H_ && ix >= 0 && ix < W_) ? ch[iy * W_ + ix] : 0.0f;
}

// A fragment: documented 16-bit A 16x32 layout. Lanes 0-15: V0..3 = K 0..7
// pairs, V4..7 = K 16..23 pairs; lanes 16-31 shifted by +8. Pairs contiguous
// in memory -> compiler merges into 2x global_load_b128 per fragment.
__device__ __forceinline__ v16bf load_a_frag(const unsigned short* __restrict__ wB,
                                             int m, int Ktot, int kbase, int h) {
    Frag16 a;
    const unsigned short* p = wB + (size_t)m * Ktot + kbase;
    #pragma unroll
    for (int i = 0; i < 8; ++i) {
        int kk = ((i < 4) ? (2 * i) : (16 + 2 * (i - 4))) + 8 * h;
        a.u[i] = *(const unsigned int*)(p + kk);
    }
    return a.v;
}

// B fragment: 32x16, lanes 0-15 K 0..7,16..23, lanes 16-31 K 8..15,24..31, K-pair per VGPR.
__device__ __forceinline__ v16bf load_b_frag(const unsigned short* __restrict__ Bsh,
                                             int n, int ks, int h) {
    Frag16 b;
    const unsigned short* p = Bsh + n * BK_STRIDE + ks + 8 * h;
    #pragma unroll
    for (int i = 0; i < 8; ++i)
        b.u[i] = *(const unsigned int*)(p + ((i < 4) ? (2 * i) : (16 + 2 * (i - 4))));
    return b.v;
}

// ---------------------------------------------------------------------------
// Generic 3x3 conv, stride 1, pad 1, as implicit GEMM on WMMA bf16.
//   in = [cat(in1, in2)] split at Cin1 (Cin1 == Cin -> no concat)
//   wB = bf16 weights, row-major [CoutPad][Cin*9] (rows >= Cout zero-padded)
// Block: 256 thr = 8 waves; tile M=64 x N=64; waves: 4(M) x 2(N), 2 acc each.
// ---------------------------------------------------------------------------
__global__ __launch_bounds__(256)
void conv3x3_wmma_kernel(const float* __restrict__ in1, const float* __restrict__ in2,
                         int Cin1, int Cin,
                         const unsigned short* __restrict__ wB,
                         const float* __restrict__ bias,
                         float* __restrict__ out, int Cout,
                         int act, const float* __restrict__ resid)
{
    __shared__ __align__(16) unsigned short Bsh[TILE_N * BK_STRIDE];

    const int tidx = threadIdx.x;
    const int lane = tidx & 31;
    const int wave = tidx >> 5;
    const int wm   = wave & 3;        // M sub-tile of this wave (0..3)
    const int wn   = wave >> 2;       // N half of this wave (0..1)
    const int h    = lane >> 4;
    const int ln   = lane & 15;

    const int tileid = blockIdx.x;
    const int xt = tileid % (W_ / TILE_N);
    const int y  = (tileid / (W_ / TILE_N)) % H_;
    const int b  = tileid / ((W_ / TILE_N) * H_);
    const int x0 = xt * TILE_N;

    const int m0w  = blockIdx.y * 64 + wm * 16;
    const int Ktot = Cin * 9;
    const int Cin2 = Cin - Cin1;

    const int n0a = wn * 32;
    const int n0b = wn * 32 + 16;

    v8f acc0 = {};
    v8f acc1 = {};

    for (int kp = 0; kp < Ktot; kp += KPANEL) {
        __syncthreads();
        // ---- build im2col K-panel (KPANEL x TILE_N) in LDS, fp32 -> bf16 ----
        #pragma unroll
        for (int r = 0; r < KPANEL * TILE_N / 256; ++r) {   // 24 elems/thread
            int idx = tidx + 256 * r;
            int kl  = idx >> 6;              // 0..95 (K within panel)
            int n   = idx & 63;              // 0..63 (spatial)
            int K   = kp + kl;
            int cin = K / 9;
            int k9  = K - cin * 9;
            int yy  = y + (k9 / 3) - 1;
            int xx  = x0 + n + (k9 % 3) - 1;
            float v = 0.0f;
            if (yy >= 0 && yy < H_ && xx >= 0 && xx < W_) {
                const float* src = (cin < Cin1)
                    ? in1 + (size_t)(b * Cin1 + cin) * (size_t)HW_
                    : in2 + (size_t)(b * Cin2 + (cin - Cin1)) * (size_t)HW_;
                v = src[yy * W_ + xx];
            }
            Bsh[n * BK_STRIDE + kl] = f2bf(v);
        }
        __syncthreads();

        if (kp + KPANEL < Ktot)   // pull next weight panel toward L2/L0
            __builtin_prefetch(wB + (size_t)(m0w + ln) * Ktot + kp + KPANEL, 0, 1);

        #pragma unroll
        for (int s = 0; s < KPANEL / KSTEP; ++s) {          // 3 WMMA k-steps
            v16bf a  = load_a_frag(wB, m0w + ln, Ktot, kp + s * KSTEP, h);
            v16bf b0 = load_b_frag(Bsh, n0a + ln, s * KSTEP, h);
            v16bf b1 = load_b_frag(Bsh, n0b + ln, s * KSTEP, h);
            acc0 = __builtin_amdgcn_wmma_f32_16x16x32_bf16(false, a, false, b0,
                                                           (short)0, acc0, false, false);
            acc1 = __builtin_amdgcn_wmma_f32_16x16x32_bf16(false, a, false, b1,
                                                           (short)0, acc1, false, false);
        }
    }

    // ---- epilogue: bias + optional lrelu + optional residual, f32 store ----
    #pragma unroll
    for (int v = 0; v < 8; ++v) {
        int m = m0w + v + 8 * h;              // C/D layout: VGPR v -> M = v + 8*half
        if (m < Cout) {
            float bv = bias[m];
            {
                int x = x0 + n0a + ln;
                float val = acc0[v] + bv;
                if (act) val = (val >= 0.0f) ? val : 0.1f * val;
                size_t oi = ((size_t)(b * Cout + m) * H_ + y) * W_ + x;
                if (resid) val += resid[oi];
                out[oi] = val;
            }
            {
                int x = x0 + n0b + ln;
                float val = acc1[v] + bv;
                if (act) val = (val >= 0.0f) ? val : 0.1f * val;
                size_t oi = ((size_t)(b * Cout + m) * H_ + y) * W_ + x;
                if (resid) val += resid[oi];
                out[oi] = val;
            }
        }
    }
}

// ---------------------------------------------------------------------------
// Modulated deformable conv v2 (3x3, dg=8) as implicit GEMM on WMMA bf16.
// om = raw conv output [B, 216, H, W]: ch 0..71 = dy, 72..143 = dx,
// 144..215 = mask logits (sigmoid applied here). The B panel is *generated*
// by bilinear sampling (zero OOB = reference valid-mask) and modulation,
// then fed to the same WMMA pipeline. Cin = Cout = 64, K = 576.
// ---------------------------------------------------------------------------
__global__ __launch_bounds__(256)
void mdcn_wmma_kernel(const float* __restrict__ xin,
                      const float* __restrict__ om,
                      const unsigned short* __restrict__ wB,
                      const float* __restrict__ bias,
                      float* __restrict__ out, int act)
{
    __shared__ __align__(16) unsigned short Bsh[TILE_N * BK_STRIDE];
    __shared__ float SPy[TILE_N * 72];
    __shared__ float SPx[TILE_N * 72];
    __shared__ float SM [TILE_N * 72];

    const int tidx = threadIdx.x;
    const int lane = tidx & 31;
    const int wave = tidx >> 5;
    const int wm   = wave & 3;
    const int wn   = wave >> 2;
    const int h    = lane >> 4;
    const int ln   = lane & 15;

    const int tileid = blockIdx.x;
    const int xt = tileid % (W_ / TILE_N);
    const int y  = (tileid / (W_ / TILE_N)) % H_;
    const int b  = tileid / ((W_ / TILE_N) * H_);
    const int x0 = xt * TILE_N;

    const int m0w  = wm * 16;
    const int Ktot = C_ * 9;          // 576

    const int n0a = wn * 32;
    const int n0b = wn * 32 + 16;

    // ---- precompute sample coords + sigmoid(mask) for this tile ----
    for (int i = tidx; i < TILE_N * 72; i += 256) {
        int n = i / 72;
        int r = i - n * 72;           // g*9 + k
        int g = r / 9;
        int k = r - g * 9;
        int x = x0 + n;
        size_t base = (size_t)b * OM_ * HW_ + (size_t)y * W_ + x;
        float oy = om[base + (size_t)(g * 9 + k) * HW_];
        float ox = om[base + (size_t)(72 + g * 9 + k) * HW_];
        float mr = om[base + (size_t)(144 + g * 9 + k) * HW_];
        SPy[i] = (float)y + (float)(k / 3 - 1) + oy;
        SPx[i] = (float)x + (float)(k % 3 - 1) + ox;
        SM[i]  = 1.0f / (1.0f + expf(-mr));
    }
    __syncthreads();

    v8f acc0 = {};
    v8f acc1 = {};

    for (int kp = 0; kp < Ktot; kp += KPANEL) {
        __syncthreads();
        // ---- generate modulated bilinear-sampled K panel in LDS ----
        #pragma unroll
        for (int r = 0; r < KPANEL * TILE_N / 256; ++r) {
            int idx = tidx + 256 * r;
            int kl  = idx >> 6;
            int n   = idx & 63;
            int K   = kp + kl;
            int cin = K / 9;
            int k9  = K - cin * 9;
            int g   = cin >> 3;                    // Cg = 8
            int t   = n * 72 + g * 9 + k9;
            float py = SPy[t], px = SPx[t], mm = SM[t];
            float fy = floorf(py), fx = floorf(px);
            int   iy = (int)fy,    ix = (int)fx;
            float wy = py - fy,    wx = px - fx;
            const float* ch = xin + (size_t)(b * C_ + cin) * (size_t)HW_;
            float v00 = ld_guard(ch, iy,     ix    );
            float v01 = ld_guard(ch, iy,     ix + 1);
            float v10 = ld_guard(ch, iy + 1, ix    );
            float v11 = ld_guard(ch, iy + 1, ix + 1);
            float v = (v00 * (1.0f - wx) + v01 * wx) * (1.0f - wy)
                    + (v10 * (1.0f - wx) + v11 * wx) * wy;
            Bsh[n * BK_STRIDE + kl] = f2bf(v * mm);
        }
        __syncthreads();

        if (kp + KPANEL < Ktot)
            __builtin_prefetch(wB + (size_t)(m0w + ln) * Ktot + kp + KPANEL, 0, 1);

        #pragma unroll
        for (int s = 0; s < KPANEL / KSTEP; ++s) {
            v16bf a  = load_a_frag(wB, m0w + ln, Ktot, kp + s * KSTEP, h);
            v16bf b0 = load_b_frag(Bsh, n0a + ln, s * KSTEP, h);
            v16bf b1 = load_b_frag(Bsh, n0b + ln, s * KSTEP, h);
            acc0 = __builtin_amdgcn_wmma_f32_16x16x32_bf16(false, a, false, b0,
                                                           (short)0, acc0, false, false);
            acc1 = __builtin_amdgcn_wmma_f32_16x16x32_bf16(false, a, false, b1,
                                                           (short)0, acc1, false, false);
        }
    }

    #pragma unroll
    for (int v = 0; v < 8; ++v) {
        int m = m0w + v + 8 * h;
        float bv = bias[m];
        {
            int x = x0 + n0a + ln;
            float val = acc0[v] + bv;
            if (act) val = (val >= 0.0f) ? val : 0.1f * val;
            out[((size_t)(b * C_ + m) * H_ + y) * W_ + x] = val;
        }
        {
            int x = x0 + n0b + ln;
            float val = acc1[v] + bv;
            if (act) val = (val >= 0.0f) ? val : 0.1f * val;
            out[((size_t)(b * C_ + m) * H_ + y) * W_ + x] = val;
        }
    }
}

// ---------------------------------------------------------------------------
// corr = sum_c emb*emb_ref ; x2 = ref * corr   (temporal-attention epilogue)
// ---------------------------------------------------------------------------
__global__ __launch_bounds__(256)
void corr_mul_kernel(const float* __restrict__ emb, const float* __restrict__ embref,
                     const float* __restrict__ ref, float* __restrict__ x2)
{
    int i = blockIdx.x * 256 + threadIdx.x;       // over B*H*W
    if (i >= B_ * HW_) return;
    int b = i / HW_;
    int p = i - b * HW_;
    size_t base = (size_t)b * C_ * HW_ + p;
    float s = 0.0f;
    for (int c = 0; c < C_; ++c)
        s += emb[base + (size_t)c * HW_] * embref[base + (size_t)c * HW_];
    for (int c = 0; c < C_; ++c) {
        size_t o = base + (size_t)c * HW_;
        x2[o] = ref[o] * s;
    }
}

// fp32 -> bf16 weight conversion with zero padding (for 216 -> 256 row pad)
__global__ __launch_bounds__(256)
void cvt_bf16_kernel(const float* __restrict__ src, unsigned short* __restrict__ dst,
                     int nsrc, int ndst)
{
    int i = blockIdx.x * 256 + threadIdx.x;
    if (i < ndst) dst[i] = (i < nsrc) ? f2bf(src[i]) : (unsigned short)0;
}

// ---------------------------------------------------------------------------
extern "C" void kernel_launch(void* const* d_in, const int* in_sizes, int n_in,
                              void* d_out, int out_size, void* d_ws, size_t ws_size,
                              hipStream_t stream)
{
    (void)in_sizes; (void)n_in; (void)out_size; (void)ws_size;

    const float* ref   = (const float*)d_in[0];
    const float* neb   = (const float*)d_in[1];
    const float* w_oc1 = (const float*)d_in[2];  const float* b_oc1 = (const float*)d_in[3];
    const float* w_om1 = (const float*)d_in[4];  const float* b_om1 = (const float*)d_in[5];
    const float* w_dc1 = (const float*)d_in[6];  const float* b_dc1 = (const float*)d_in[7];
    const float* w_fc  = (const float*)d_in[8];  const float* b_fc  = (const float*)d_in[9];
    const float* w_cs1 = (const float*)d_in[10]; const float* b_cs1 = (const float*)d_in[11];
    const float* w_cs2 = (const float*)d_in[12]; const float* b_cs2 = (const float*)d_in[13];
    const float* w_om2 = (const float*)d_in[14]; const float* b_om2 = (const float*)d_in[15];
    const float* w_dc2 = (const float*)d_in[16]; const float* b_dc2 = (const float*)d_in[17];
    const float* w_ta1 = (const float*)d_in[18]; const float* b_ta1 = (const float*)d_in[19];
    const float* w_ta2 = (const float*)d_in[20]; const float* b_ta2 = (const float*)d_in[21];
    const float* w_ff  = (const float*)d_in[22]; const float* b_ff  = (const float*)d_in[23];
    float* outp = (float*)d_out;

    // ---- workspace layout: bf16 weights, then f32 activation buffers ----
    unsigned short* wsu = (unsigned short*)d_ws;
    size_t woff = 0;
    auto walloc = [&](size_t n) { unsigned short* p = wsu + woff; woff += n; return p; };
    const int K1 = 64 * 9, K2 = 128 * 9;
    unsigned short* wb_oc1 = walloc((size_t)64  * K2);
    unsigned short* wb_om1 = walloc((size_t)256 * K1);   // 216 rows padded to 256
    unsigned short* wb_dc1 = walloc((size_t)64  * K1);
    unsigned short* wb_fc  = walloc((size_t)64  * K1);
    unsigned short* wb_cs1 = walloc((size_t)64  * K2);
    unsigned short* wb_cs2 = walloc((size_t)64  * K1);
    unsigned short* wb_om2 = walloc((size_t)256 * K1);
    unsigned short* wb_dc2 = walloc((size_t)64  * K1);
    unsigned short* wb_ta1 = walloc((size_t)64  * K1);
    unsigned short* wb_ta2 = walloc((size_t)64  * K1);
    unsigned short* wb_ff  = walloc((size_t)64  * K1);

    size_t wbytes = (woff * sizeof(unsigned short) + 255) & ~(size_t)255;
    const size_t A64 = (size_t)B_ * C_ * HW_;            // 14,745,600 f32
    float* bufA  = (float*)((char*)d_ws + wbytes);
    float* bufB  = bufA + A64;
    float* bufC  = bufB + A64;
    float* bufOM = bufC + A64;                           // B * 216 * H * W f32

    auto cvt = [&](const float* s, unsigned short* d, int ns, int nd) {
        cvt_bf16_kernel<<<(nd + 255) / 256, 256, 0, stream>>>(s, d, ns, nd);
    };
    cvt(w_oc1, wb_oc1, 64 * K2, 64 * K2);
    cvt(w_om1, wb_om1, OM_ * K1, 256 * K1);
    cvt(w_dc1, wb_dc1, 64 * K1, 64 * K1);
    cvt(w_fc,  wb_fc,  64 * K1, 64 * K1);
    cvt(w_cs1, wb_cs1, 64 * K2, 64 * K2);
    cvt(w_cs2, wb_cs2, 64 * K1, 64 * K1);
    cvt(w_om2, wb_om2, OM_ * K1, 256 * K1);
    cvt(w_dc2, wb_dc2, 64 * K1, 64 * K1);
    cvt(w_ta1, wb_ta1, 64 * K1, 64 * K1);
    cvt(w_ta2, wb_ta2, 64 * K1, 64 * K1);
    cvt(w_ff,  wb_ff,  64 * K1, 64 * K1);

    auto conv = [&](const float* i1, const float* i2, int c1, int c,
                    const unsigned short* wb, const float* bias, float* o,
                    int cout, int act, const float* resid) {
        dim3 grid(GRID_X, (cout + 63) / 64);
        conv3x3_wmma_kernel<<<grid, 256, 0, stream>>>(i1, i2, c1, c, wb, bias, o,
                                                      cout, act, resid);
    };
    auto mdcn = [&](const float* x, const float* om, const unsigned short* wb,
                    const float* bias, float* o, int act) {
        mdcn_wmma_kernel<<<dim3(GRID_X, 1), 256, 0, stream>>>(x, om, wb, bias, o, act);
    };

    // 1. off = lrelu(conv(cat(neb, ref), oc1))
    conv(neb, ref, 64, 128, wb_oc1, b_oc1, bufA, 64, 1, nullptr);
    // 2. om1 = conv(off, om1)  [216 ch raw offset/mask]
    conv(bufA, nullptr, 64, 64, wb_om1, b_om1, bufOM, OM_, 0, nullptr);
    // 3. feat = mdcn(neb, om1, dcn1)
    mdcn(neb, bufOM, wb_dc1, b_dc1, bufB, 0);
    // 4. feat = lrelu(conv(feat, fc))
    conv(bufB, nullptr, 64, 64, wb_fc, b_fc, bufC, 64, 1, nullptr);
    // 5. t = lrelu(conv(cat(feat, ref), cas1))
    conv(bufC, ref, 64, 128, wb_cs1, b_cs1, bufA, 64, 1, nullptr);
    // 6. off2 = lrelu(conv(t, cas2))
    conv(bufA, nullptr, 64, 64, wb_cs2, b_cs2, bufB, 64, 1, nullptr);
    // 7. om2 = conv(off2, om2)
    conv(bufB, nullptr, 64, 64, wb_om2, b_om2, bufOM, OM_, 0, nullptr);
    // 8. feat2 = lrelu(mdcn(feat, om2, dcn2))
    mdcn(bufC, bufOM, wb_dc2, b_dc2, bufA, 1);
    // 9. emb_ref = conv(ref, ta1)
    conv(ref, nullptr, 64, 64, wb_ta1, b_ta1, bufB, 64, 0, nullptr);
    // 10. emb = conv(feat2, ta2)
    conv(bufA, nullptr, 64, 64, wb_ta2, b_ta2, bufC, 64, 0, nullptr);
    // 11. x2 = ref * sum_c(emb * emb_ref)
    corr_mul_kernel<<<(B_ * HW_ + 255) / 256, 256, 0, stream>>>(bufC, bufB, ref, bufA);
    // 12. out = lrelu(conv(x2, ff)) + ref
    conv(bufA, nullptr, 64, 64, wb_ff, b_ff, outp, 64, 1, ref);
}
